// MultiHeadAttention_50826642981110
// MI455X (gfx1250) — compile-verified, matrix operand order checked by reading, not run on hardware
//
#include <hip/hip_runtime.h>
#include <hip/hip_bf16.h>

// MI455X / gfx1250 multi-head attention forward.
// All matmuls via v_wmma_f32_16x16x32_f16 (f16 in, f32 acc).
// GEMM B-panels staged into LDS by the Tensor Data Mover (double-buffered),
// with a cooperative-copy fallback if the TDM builtin is unavailable.
// Workspace requirement: 64 MB.

typedef __attribute__((ext_vector_type(16))) _Float16     v16h;
typedef __attribute__((ext_vector_type(8)))  _Float16     v8h;
typedef __attribute__((ext_vector_type(8)))  float        v8f;
typedef __attribute__((ext_vector_type(4)))  unsigned int v4u;
typedef __attribute__((ext_vector_type(8)))  int          v8i;
typedef __attribute__((ext_vector_type(4)))  int          v4i;

#define D_MODEL 1024
#define SEQ     2048
#define NB      2
#define HEADS   16
#define DK      64
#define NTOK    (NB * SEQ)   // 4096 rows

#if defined(__has_builtin)
#if __has_builtin(__builtin_amdgcn_tensor_load_to_lds) && \
    __has_builtin(__builtin_amdgcn_s_wait_tensorcnt)
#define USE_TDM 1
#endif
#endif
#ifndef USE_TDM
#define USE_TDM 0
#endif

// ---------------------------------------------------------------------------
// Fragment loader for the CDNA5 16-bit A/B WMMA layout (16x32 A, 32x16 B).
// Lane l (l<16) holds row/col idx=l, K = {0..7, 16..23}; lane l+16 holds the
// same idx with K = {8..15, 24..31}.  Two 16-byte loads per lane.
// Works for global or LDS source pointers.
// ---------------------------------------------------------------------------
__device__ inline v16h load_frag16(const _Float16* p, int ld, int k0) {
  const int lane = threadIdx.x & 31;
  const int idx  = lane & 15;
  const int koff = k0 + ((lane >> 4) << 3);
  const _Float16* q = p + (size_t)idx * ld + koff;
  v8h lo = *(const v8h*)(q);
  v8h hi = *(const v8h*)(q + 16);
  v16h r;
#pragma unroll
  for (int e = 0; e < 8; ++e) { r[e] = lo[e]; r[e + 8] = hi[e]; }
  return r;
}

__device__ inline v8f wmma_f16(v16h a, v16h b, v8f c) {
  return __builtin_amdgcn_wmma_f32_16x16x32_f16(
      /*neg_a=*/false, a, /*neg_b=*/false, b,
      /*c_mod=*/(short)0, c, /*reuse_a=*/false, /*reuse_b=*/false);
}

// ---------------------------------------------------------------------------
// TDM: stage a 64-row x 32-half panel of W (row-major, row stride D_MODEL
// halves) into LDS at lds_dst, rows packed contiguously (32 halves = 64B/row).
// D# built per CDNA5 ISA ch.8: data_size=2B, tile_dim0=32 (contiguous K),
// tile_dim1=64 (W rows), tensor_dim0_stride=D_MODEL, descriptor type=2.
// ---------------------------------------------------------------------------
__device__ inline void tdm_stage_panel(_Float16* lds_dst, const _Float16* gsrc) {
#if USE_TDM
  const unsigned long long ga = (unsigned long long)(uintptr_t)gsrc;
  const unsigned int ldsoff = (unsigned int)(uintptr_t)lds_dst; // addr[31:0] = LDS offset
  const v4u g0 = { 1u,                                   // count=1, user mode
                   ldsoff,                               // lds_addr
                   (unsigned int)(ga & 0xffffffffu),     // global_addr[31:0]
                   (unsigned int)((ga >> 32) & 0x1ffffffu) | (2u << 30) }; // [56:32] | type=2
  const v8i g1 = { 0x00010000,                 // data_size=1 (2 bytes); mask=0
                   (int)(D_MODEL << 16),       // tensor_dim0[15:0] in [31:16]
                   (int)(D_MODEL << 16),       // tensor_dim0 hi=0 | tensor_dim1[15:0]
                   (int)(32u << 16),           // tensor_dim1 hi=0 | tile_dim0=32
                   64,                         // tile_dim1=64 | tile_dim2=0
                   D_MODEL,                    // tensor_dim0_stride[31:0]
                   0, 0 };
  const v4i g2 = { 0, 0, 0, 0 };
  const v4i g3 = { 0, 0, 0, 0 };
#if defined(__clang_major__) && (__clang_major__ >= 23)
  const v8i g4 = { 0, 0, 0, 0, 0, 0, 0, 0 };
  __builtin_amdgcn_tensor_load_to_lds(g0, g1, g2, g3, g4, 0);
#else
  __builtin_amdgcn_tensor_load_to_lds(g0, g1, g2, g3, 0);
#endif
#else
  (void)lds_dst; (void)gsrc;
#endif
}

// Fallback cooperative panel copy (all 128 threads): 64 rows x 64B.
__device__ inline void copy_panel(_Float16* dst, const _Float16* src) {
  const int t = threadIdx.x;
#pragma unroll
  for (int i = 0; i < 2; ++i) {
    const int ch  = t + i * 128;          // 256 chunks of 16B
    const int c   = ch >> 2;
    const int seg = ch & 3;
    *(v8h*)(dst + c * 32 + seg * 8) =
        *(const v8h*)(src + (size_t)c * D_MODEL + seg * 8);
  }
}

// ---------------------------------------------------------------------------
// f32 -> f16 conversion (n always a multiple of 1024 here)
// ---------------------------------------------------------------------------
__global__ void cvt_f32_to_f16(const float* __restrict__ in,
                               _Float16* __restrict__ out, int n) {
  int i = (blockIdx.x * blockDim.x + threadIdx.x) * 4;
  if (i + 3 < n) {
    float4 v = *(const float4*)(in + i);
    out[i + 0] = (_Float16)v.x;
    out[i + 1] = (_Float16)v.y;
    out[i + 2] = (_Float16)v.z;
    out[i + 3] = (_Float16)v.w;
  }
}

// ---------------------------------------------------------------------------
// WMMA GEMM:  out[m,n] = sum_k A[m,k] * W[n,k] + bias[n]
// Block = 4 waves; block tile = 128 rows x 64 cols; wave tile = 32 x 64
// (2 A-frags x 4 B-frags -> 8 WMMAs per 32-wide K-step).
// The 64x32 B panel is shared by all waves: staged in LDS, double-buffered,
// loaded by the TDM (wave 0 issues; s_wait_tensorcnt + barrier to consume).
// Store modes:
//   0: Q-proj -> f16 [B*H, S, DK], scaled by 1/sqrt(DK)
//   1: K-proj -> f16 [B*H, S, DK]
//   2: V-proj -> f16 [B*H, DK, S] (transposed so P@V B-frags are contiguous)
//   3: O-proj -> f32 [NTOK, D_MODEL] (final output)
// ---------------------------------------------------------------------------
__global__ __launch_bounds__(128) void gemm_wmma(const _Float16* __restrict__ A,
                                                 const _Float16* __restrict__ W,
                                                 const float* __restrict__ bias,
                                                 void* __restrict__ out, int mode) {
  __shared__ __align__(16) _Float16 sB[2][64][32];   // 2 x 4 KB double buffer

  const int wave = threadIdx.x >> 5;
  const int lane = threadIdx.x & 31;
  const int row0 = blockIdx.x * 128 + wave * 32;
  const int col0 = blockIdx.y * 64;

  v8f acc[2][4] = {};

  // Prologue: stage panel for k0 = 0 into buffer 0.
#if USE_TDM
  if (wave == 0) {
    tdm_stage_panel(&sB[0][0][0], W + (size_t)col0 * D_MODEL);
    __builtin_amdgcn_s_wait_tensorcnt(0);
  }
#else
  copy_panel(&sB[0][0][0], W + (size_t)col0 * D_MODEL);
#endif
  __syncthreads();

  for (int kk = 0; kk < D_MODEL / 32; ++kk) {
    const int k0  = kk * 32;
    const int cur = kk & 1;

#if USE_TDM
    // Issue async TDM for the next panel while computing on the current one.
    if (wave == 0 && k0 + 32 < D_MODEL)
      tdm_stage_panel(&sB[cur ^ 1][0][0],
                      W + (size_t)col0 * D_MODEL + (k0 + 32));
#endif

    v16h af0 = load_frag16(A + (size_t)row0 * D_MODEL, D_MODEL, k0);
    v16h af1 = load_frag16(A + (size_t)(row0 + 16) * D_MODEL, D_MODEL, k0);
#pragma unroll
    for (int t = 0; t < 4; ++t) {
      v16h bf = load_frag16(&sB[cur][t * 16][0], 32, 0);
      acc[0][t] = wmma_f16(af0, bf, acc[0][t]);
      acc[1][t] = wmma_f16(af1, bf, acc[1][t]);
    }

#if USE_TDM
    if (wave == 0) __builtin_amdgcn_s_wait_tensorcnt(0);
    __syncthreads();
#else
    __syncthreads();
    if (k0 + 32 < D_MODEL)
      copy_panel(&sB[cur ^ 1][0][0], W + (size_t)col0 * D_MODEL + (k0 + 32));
    __syncthreads();
#endif
  }

  // C/D layout: VGPR r, lanes 0-15 -> row r; lanes 16-31 -> row r+8; col = lane&15
  const int rb = (lane >> 4) << 3;
  const int cl = lane & 15;
#pragma unroll
  for (int rt = 0; rt < 2; ++rt) {
#pragma unroll
    for (int t = 0; t < 4; ++t) {
#pragma unroll
      for (int r = 0; r < 8; ++r) {
        const int row = row0 + rt * 16 + r + rb;
        const int col = col0 + t * 16 + cl;
        const float val = acc[rt][t][r] + bias[col];
        const int b = row >> 11, s = row & (SEQ - 1);
        const int h = col >> 6,  d = col & (DK - 1);
        if (mode == 0) {
          ((_Float16*)out)[((size_t)(b * HEADS + h) * SEQ + s) * DK + d] =
              (_Float16)(val * 0.125f);   // fold 1/sqrt(64)
        } else if (mode == 1) {
          ((_Float16*)out)[((size_t)(b * HEADS + h) * SEQ + s) * DK + d] = (_Float16)val;
        } else if (mode == 2) {
          ((_Float16*)out)[((size_t)(b * HEADS + h) * DK + d) * SEQ + s] = (_Float16)val;
        } else {
          ((float*)out)[(size_t)row * D_MODEL + col] = val;
        }
      }
    }
  }
}

// ---------------------------------------------------------------------------
// Flash attention: one wave per 16-query tile, online softmax over key tiles
// of 32.  Q,K: [B*H, S, DK] f16 (Q pre-scaled); Vt: [B*H, DK, S] f16.
// Output O: f16 [NTOK, D_MODEL] (col = h*64 + d), input to the final GEMM.
// ---------------------------------------------------------------------------
__global__ __launch_bounds__(128) void attn_wmma(const _Float16* __restrict__ Q,
                                                 const _Float16* __restrict__ K,
                                                 const _Float16* __restrict__ Vt,
                                                 _Float16* __restrict__ O) {
  __shared__ __align__(16) float    s_sc[4][16][32];
  __shared__ __align__(16) _Float16 s_p[4][16][32];
  __shared__ float s_alpha[4][16];
  __shared__ float s_m[4][16];
  __shared__ float s_l[4][16];

  const int wave = threadIdx.x >> 5;
  const int lane = threadIdx.x & 31;
  const int gw   = blockIdx.x * 4 + wave;      // 0..4095 wave tiles
  const int qt   = gw & (SEQ / 16 - 1);        // query tile 0..127
  const int bh   = gw >> 7;                    // 0..31

  const _Float16* Qp = Q  + ((size_t)bh * SEQ + (size_t)qt * 16) * DK;
  const _Float16* Kp = K  + (size_t)bh * SEQ * DK;
  const _Float16* Vp = Vt + (size_t)bh * DK * SEQ;

  v16h qf0 = load_frag16(Qp, DK, 0);
  v16h qf1 = load_frag16(Qp, DK, 32);
  v8f oacc[4] = {};

  if (lane < 16) { s_m[wave][lane] = -3.0e38f; s_l[wave][lane] = 0.0f; }
  __syncthreads();

  const int rb = (lane >> 4) << 3;
  const int cl = lane & 15;

  for (int kt = 0; kt < SEQ; kt += 32) {
    // ---- scores tile: 16 queries x 32 keys ----
#pragma unroll
    for (int j = 0; j < 2; ++j) {
      v8f sc = {};
      v16h kf0 = load_frag16(Kp + (size_t)(kt + j * 16) * DK, DK, 0);
      v16h kf1 = load_frag16(Kp + (size_t)(kt + j * 16) * DK, DK, 32);
      sc = wmma_f16(qf0, kf0, sc);
      sc = wmma_f16(qf1, kf1, sc);
#pragma unroll
      for (int r = 0; r < 8; ++r) s_sc[wave][r + rb][j * 16 + cl] = sc[r];
    }
    __syncthreads();

    // ---- online softmax update (one row per lane 0..15) ----
    if (lane < 16) {
      const int r = lane;
      float mold = s_m[wave][r];
      float mx = mold;
#pragma unroll
      for (int t = 0; t < 32; ++t) mx = fmaxf(mx, s_sc[wave][r][t]);
      float alpha = __expf(mold - mx);
      float lsum = 0.0f;
#pragma unroll
      for (int t = 0; t < 32; ++t) {
        float p = __expf(s_sc[wave][r][t] - mx);
        lsum += p;
        s_p[wave][r][t] = (_Float16)p;
      }
      s_m[wave][r]     = mx;
      s_l[wave][r]     = s_l[wave][r] * alpha + lsum;
      s_alpha[wave][r] = alpha;
    }
    __syncthreads();

    // ---- rescale running output, then O += P @ V ----
    float al[8];
#pragma unroll
    for (int r = 0; r < 8; ++r) al[r] = s_alpha[wave][r + rb];
#pragma unroll
    for (int t = 0; t < 4; ++t)
#pragma unroll
      for (int r = 0; r < 8; ++r) oacc[t][r] *= al[r];

    v16h pf = load_frag16(&s_p[wave][0][0], 32, 0);   // LDS -> A-frag (transpose bounce)
#pragma unroll
    for (int t = 0; t < 4; ++t) {
      v16h vf = load_frag16(Vp + (size_t)(t * 16) * SEQ + kt, SEQ, 0);
      oacc[t] = wmma_f16(pf, vf, oacc[t]);
    }
    __syncthreads();
  }

  float inv[8];
#pragma unroll
  for (int r = 0; r < 8; ++r) inv[r] = 1.0f / s_l[wave][r + rb];
  const int b = bh >> 4, h = bh & (HEADS - 1);
#pragma unroll
  for (int t = 0; t < 4; ++t) {
#pragma unroll
    for (int r = 0; r < 8; ++r) {
      const int srow = qt * 16 + r + rb;
      const int col  = h * DK + t * 16 + cl;
      O[(size_t)(b * SEQ + srow) * D_MODEL + col] = (_Float16)(oacc[t][r] * inv[r]);
    }
  }
}

// ---------------------------------------------------------------------------
extern "C" void kernel_launch(void* const* d_in, const int* in_sizes, int n_in,
                              void* d_out, int out_size, void* d_ws, size_t ws_size,
                              hipStream_t stream) {
  (void)in_sizes; (void)n_in; (void)out_size; (void)ws_size;
  const float* q  = (const float*)d_in[0];
  const float* k  = (const float*)d_in[1];
  const float* v  = (const float*)d_in[2];
  const float* Wq = (const float*)d_in[3];
  const float* bq = (const float*)d_in[4];
  const float* Wk = (const float*)d_in[5];
  const float* bk = (const float*)d_in[6];
  const float* Wv = (const float*)d_in[7];
  const float* bv = (const float*)d_in[8];
  const float* Wo = (const float*)d_in[9];
  const float* bo = (const float*)d_in[10];

  const size_t NTE = (size_t)NTOK * D_MODEL;     // 4 Mi elems
  const size_t WTE = (size_t)D_MODEL * D_MODEL;  // 1 Mi elems
  _Float16* qf   = (_Float16*)d_ws;
  _Float16* kf   = qf   + NTE;
  _Float16* vf   = kf   + NTE;
  _Float16* wq16 = vf   + NTE;
  _Float16* wk16 = wq16 + WTE;
  _Float16* wv16 = wk16 + WTE;
  _Float16* wo16 = wv16 + WTE;
  _Float16* Qh   = wo16 + WTE;                   // [B*H, S, DK], pre-scaled
  _Float16* Kh   = Qh   + NTE;                   // [B*H, S, DK]
  _Float16* Vt   = Kh   + NTE;                   // [B*H, DK, S]
  _Float16* Oa   = Vt   + NTE;                   // attention output, f16
  // total = (7*NTE + 4*WTE)*2B = 64 MB

  const dim3 cb(256);
  cvt_f32_to_f16<<<dim3((unsigned)(NTE / 1024)), cb, 0, stream>>>(q, qf, (int)NTE);
  cvt_f32_to_f16<<<dim3((unsigned)(NTE / 1024)), cb, 0, stream>>>(k, kf, (int)NTE);
  cvt_f32_to_f16<<<dim3((unsigned)(NTE / 1024)), cb, 0, stream>>>(v, vf, (int)NTE);
  cvt_f32_to_f16<<<dim3((unsigned)(WTE / 1024)), cb, 0, stream>>>(Wq, wq16, (int)WTE);
  cvt_f32_to_f16<<<dim3((unsigned)(WTE / 1024)), cb, 0, stream>>>(Wk, wk16, (int)WTE);
  cvt_f32_to_f16<<<dim3((unsigned)(WTE / 1024)), cb, 0, stream>>>(Wv, wv16, (int)WTE);
  cvt_f32_to_f16<<<dim3((unsigned)(WTE / 1024)), cb, 0, stream>>>(Wo, wo16, (int)WTE);

  const dim3 gblk(128);                          // 4 waves
  const dim3 ggrid(NTOK / 128, D_MODEL / 64);    // (32, 16)
  gemm_wmma<<<ggrid, gblk, 0, stream>>>(qf, wq16, bq, Qh, 0);
  gemm_wmma<<<ggrid, gblk, 0, stream>>>(kf, wk16, bk, Kh, 1);
  gemm_wmma<<<ggrid, gblk, 0, stream>>>(vf, wv16, bv, Vt, 2);

  attn_wmma<<<dim3(NB * HEADS * (SEQ / 16) / 4), gblk, 0, stream>>>(Qh, Kh, Vt, Oa);

  gemm_wmma<<<ggrid, gblk, 0, stream>>>(Oa, wo16, bo, d_out, 3);
}
